// SingleHeadAttention_8126078124685
// MI455X (gfx1250) — compile-verified, matrix operand order checked by reading, not run on hardware
//
#include <hip/hip_runtime.h>

typedef __attribute__((ext_vector_type(16))) _Float16 v16h;
typedef __attribute__((ext_vector_type(8)))  _Float16 v8h;
typedef __attribute__((ext_vector_type(8)))  float    v8f;

#define B_SZ   64
#define NQ     1024
#define NK     1024
#define DIM    256
#define NEG_INF_V -100000000.0f

// ---- WMMA operand loaders -------------------------------------------------
// A/B operand (16x32 f16 tile slice): lane l holds row (l&15); half h=(l>>4)
// covers K elements [kbase+8h, kbase+8h+8) and [kbase+16+8h, kbase+16+8h+8).

__device__ __forceinline__ v16h load16_f16(const _Float16* __restrict__ row,
                                           int kbase, int h) {
  const v8h lo = *(const v8h*)(row + kbase + 8 * h);
  const v8h hi = *(const v8h*)(row + kbase + 16 + 8 * h);
  v16h r;
#pragma unroll
  for (int i = 0; i < 8; ++i) { r[i] = lo[i]; r[i + 8] = hi[i]; }
  return r;
}

__device__ __forceinline__ v16h load16_f32cvt(const float* __restrict__ row,
                                              int kbase, int h) {
  const float4 a0 = *(const float4*)(row + kbase + 8 * h);
  const float4 a1 = *(const float4*)(row + kbase + 8 * h + 4);
  const float4 b0 = *(const float4*)(row + kbase + 16 + 8 * h);
  const float4 b1 = *(const float4*)(row + kbase + 16 + 8 * h + 4);
  v16h r;
  r[0]=(_Float16)a0.x; r[1]=(_Float16)a0.y; r[2]=(_Float16)a0.z; r[3]=(_Float16)a0.w;
  r[4]=(_Float16)a1.x; r[5]=(_Float16)a1.y; r[6]=(_Float16)a1.z; r[7]=(_Float16)a1.w;
  r[8]=(_Float16)b0.x; r[9]=(_Float16)b0.y; r[10]=(_Float16)b0.z; r[11]=(_Float16)b0.w;
  r[12]=(_Float16)b1.x; r[13]=(_Float16)b1.y; r[14]=(_Float16)b1.z; r[15]=(_Float16)b1.w;
  return r;
}

// ---- Kernel 1: transpose + convert weights (fold 1/sqrt(D) into Wq) -------
__global__ void sha_wt_kernel(const float* __restrict__ wq,
                              const float* __restrict__ wk,
                              _Float16* __restrict__ wq_t,
                              _Float16* __restrict__ wk_t) {
  const int k = threadIdx.x;          // 0..255
  const int n = blockIdx.x;           // 0..255
  wq_t[n * DIM + k] = (_Float16)(wq[k * DIM + n] * 0.0625f);  // 1/sqrt(256)
  wk_t[n * DIM + k] = (_Float16)(wk[k * DIM + n]);
}

// ---- Kernel 2: projection  dst[b,m,:] = src[b,m,:] @ W  (f16 out) ---------
// one wave per 16-row tile; grid.x * 8 waves = B * NQ / 16 = 4096
__global__ __launch_bounds__(256)
void sha_proj_kernel(const float* __restrict__ src,
                     const _Float16* __restrict__ wt,   // [N=256][K=256]
                     _Float16* __restrict__ dst) {
  const int lane = threadIdx.x & 31;
  const int wave = threadIdx.x >> 5;
  const int tile = blockIdx.x * 8 + wave;   // 0..4095
  const int b  = tile >> 6;
  const int qt = tile & 63;
  const int m  = lane & 15;
  const int h  = lane >> 4;

  const float* arow = src + (size_t)(b * NQ + qt * 16 + m) * DIM;
  v16h a[8];
#pragma unroll
  for (int c = 0; c < 8; ++c) a[c] = load16_f32cvt(arow, 32 * c, h);

#pragma unroll 1
  for (int nt = 0; nt < 16; ++nt) {
    const _Float16* brow = wt + (size_t)(nt * 16 + m) * DIM;  // lane n = m
    v8f acc0 = {}, acc1 = {};
#pragma unroll
    for (int c = 0; c < 4; ++c) {
      v16h bv0 = load16_f16(brow, 32 * c, h);
      v16h bv1 = load16_f16(brow, 32 * (c + 4), h);
      acc0 = __builtin_amdgcn_wmma_f32_16x16x32_f16(
          false, a[c], false, bv0, (short)0, acc0, false, false);
      acc1 = __builtin_amdgcn_wmma_f32_16x16x32_f16(
          false, a[c + 4], false, bv1, (short)0, acc1, false, false);
    }
#pragma unroll
    for (int v = 0; v < 8; ++v) {
      const int orow = b * NQ + qt * 16 + v + 8 * h;   // C: row = v + 8h
      dst[(size_t)orow * DIM + nt * 16 + m] = (_Float16)(acc0[v] + acc1[v]);
    }
  }
}

// ---- Kernel 3: scores + tanh clip + mask + log_softmax --------------------
// 2 waves per block share one 16x1024 score tile in 64KB LDS.
// Each wave computes half the K-tiles, then does log-softmax on 8 rows.
// Mask is prefetched during the WMMA phase and applied (coalesced int4
// loads) in the row pass. Output written exactly once.
__global__ __launch_bounds__(64)
void sha_attn_kernel(const _Float16* __restrict__ Q16,
                     const _Float16* __restrict__ K16,
                     const int* __restrict__ mask,
                     float* __restrict__ out) {
  __shared__ float smem[16 * NK];     // 64 KB
  const int lane = threadIdx.x & 31;
  const int wave = threadIdx.x >> 5;  // 0 or 1
  const int tile = blockIdx.x;        // 0..4095
  const int b  = tile >> 6;
  const int qt = tile & 63;
  const int m  = lane & 15;
  const int h  = lane >> 4;

  const _Float16* arow = Q16 + (size_t)(b * NQ + qt * 16 + m) * DIM;
  v16h a[8];
#pragma unroll
  for (int c = 0; c < 8; ++c) a[c] = load16_f16(arow, 32 * c, h);

  const _Float16* kmat  = K16 + (size_t)b * NK * DIM;
  const int*      mbase = mask + (size_t)(b * NQ + qt * 16) * NK;

  const int nt0 = wave * (NK / 32);   // 32 K-tiles per wave
#pragma unroll 1
  for (int nt = nt0; nt < nt0 + NK / 32; ++nt) {
    // Prefetch the mask slice for the post-barrier pass (1KB per iteration;
    // 64 iterations across both waves cover the whole 64KB block slice).
    __builtin_prefetch((const char*)mbase + nt * 1024 + (lane & 7) * 128, 0, 1);

    const _Float16* brow = kmat + (size_t)(nt * 16 + m) * DIM;  // lane n = m
    v8f acc0 = {}, acc1 = {};
#pragma unroll
    for (int c = 0; c < 4; ++c) {
      v16h bv0 = load16_f16(brow, 32 * c, h);
      v16h bv1 = load16_f16(brow, 32 * (c + 4), h);
      acc0 = __builtin_amdgcn_wmma_f32_16x16x32_f16(
          false, a[c], false, bv0, (short)0, acc0, false, false);
      acc1 = __builtin_amdgcn_wmma_f32_16x16x32_f16(
          false, a[c + 4], false, bv1, (short)0, acc1, false, false);
    }
#pragma unroll
    for (int v = 0; v < 8; ++v) {
      const int rr = v + 8 * h;       // row within tile
      const int cc = nt * 16 + m;     // column
      smem[rr * NK + cc] = 10.0f * tanhf(acc0[v] + acc1[v]);
    }
  }
  __syncthreads();

  float* obase = out + (size_t)(b * NQ + qt * 16) * NK;
  const int mm0 = wave * 8;           // 8 rows per wave
#pragma unroll 1
  for (int mm = mm0; mm < mm0 + 8; ++mm) {
    float vals[32];
    float vmax = -3.4e38f;
#pragma unroll
    for (int i = 0; i < 8; ++i) {
      const int off = mm * NK + i * 128 + lane * 4;
      const float4 t = *(const float4*)&smem[off];
      const int4  mk = *(const int4*)&mbase[off];
      vals[4*i+0] = (mk.x == 1) ? NEG_INF_V : t.x;
      vals[4*i+1] = (mk.y == 1) ? NEG_INF_V : t.y;
      vals[4*i+2] = (mk.z == 1) ? NEG_INF_V : t.z;
      vals[4*i+3] = (mk.w == 1) ? NEG_INF_V : t.w;
      vmax = fmaxf(vmax, fmaxf(fmaxf(vals[4*i+0], vals[4*i+1]),
                               fmaxf(vals[4*i+2], vals[4*i+3])));
    }
#pragma unroll
    for (int off = 16; off >= 1; off >>= 1)
      vmax = fmaxf(vmax, __shfl_xor(vmax, off, 32));

    float sum = 0.0f;
#pragma unroll
    for (int i = 0; i < 32; ++i) sum += __expf(vals[i] - vmax);
#pragma unroll
    for (int off = 16; off >= 1; off >>= 1)
      sum += __shfl_xor(sum, off, 32);

    const float lse = vmax + __logf(sum);
#pragma unroll
    for (int i = 0; i < 8; ++i) {
      float4 o;
      o.x = vals[4*i+0] - lse; o.y = vals[4*i+1] - lse;
      o.z = vals[4*i+2] - lse; o.w = vals[4*i+3] - lse;
      *(float4*)&obase[mm * NK + i * 128 + lane * 4] = o;
    }
  }
}

// ---- Launch ---------------------------------------------------------------
extern "C" void kernel_launch(void* const* d_in, const int* in_sizes, int n_in,
                              void* d_out, int out_size, void* d_ws, size_t ws_size,
                              hipStream_t stream) {
  const float* q    = (const float*)d_in[0];
  const float* k    = (const float*)d_in[1];
  const float* wq   = (const float*)d_in[2];
  const float* wk   = (const float*)d_in[3];
  const int*   mask = (const int*)d_in[4];
  float* out = (float*)d_out;

  _Float16* Q16 = (_Float16*)d_ws;
  _Float16* K16 = Q16 + (size_t)B_SZ * NQ * DIM;
  _Float16* WqT = K16 + (size_t)B_SZ * NK * DIM;
  _Float16* WkT = WqT + (size_t)DIM * DIM;

  sha_wt_kernel<<<DIM, DIM, 0, stream>>>(wq, wk, WqT, WkT);
  sha_proj_kernel<<<(B_SZ * NQ / 16) / 8, 256, 0, stream>>>(q, WqT, Q16);
  sha_proj_kernel<<<(B_SZ * NK / 16) / 8, 256, 0, stream>>>(k, WkT, K16);
  sha_attn_kernel<<<B_SZ * (NQ / 16), 64, 0, stream>>>(Q16, K16, mask, out);
}
